// Quant_Linear_Int_1932735283616
// MI455X (gfx1250) — 8-bit matrix operand order repaired and checked by reading, compiled, not run on hardware
//
#include <hip/hip_runtime.h>
#include <cstdint>


#define T_DIM   8192
#define DIN     2048
#define DOUT    2048
#define NLEV    256

typedef __attribute__((ext_vector_type(8))) int  v8i;
typedef __attribute__((ext_vector_type(4))) int  v4i;

__global__ __launch_bounds__(256)
void k_xmax_partial(const float* __restrict__ x, float* __restrict__ partial, int n) {
    __shared__ float sm[256];
    const int tid = threadIdx.x;
    int gid = blockIdx.x * 256 + tid;
    const int stride = gridDim.x * 256;
    float m = 0.0f;
    for (int i = gid; i < n; i += stride) m = fmaxf(m, x[i]);
    sm[tid] = m;
    __syncthreads();
    for (int s = 128; s > 0; s >>= 1) {
        if (tid < s) sm[tid] = fmaxf(sm[tid], sm[tid + s]);
        __syncthreads();
    }
    if (tid == 0) partial[blockIdx.x] = sm[0];
}

__global__ __launch_bounds__(256)
void k_xscale(const float* __restrict__ partial, float* __restrict__ scal) {
    __shared__ float sm[256];
    const int tid = threadIdx.x;
    sm[tid] = partial[tid];
    __syncthreads();
    for (int s = 128; s > 0; s >>= 1) {
        if (tid < s) sm[tid] = fmaxf(sm[tid], sm[tid + s]);
        __syncthreads();
    }
    if (tid == 0) {
        float xm = fmaxf(sm[0], 1e-8f);
        float sx = (float)(NLEV - 1) / xm;
        scal[0] = sx;
        scal[1] = 1.0f / sx;
    }
}

__global__ __launch_bounds__(256)
void k_wquant(const float* __restrict__ w, const float* __restrict__ bias,
              const float* __restrict__ scal,
              float* __restrict__ zp, float* __restrict__ isw,
              float* __restrict__ qbf, unsigned char* __restrict__ qw) {
    __shared__ float smin[256], smax[256];
    __shared__ float s_sw, s_zp;
    const int o = blockIdx.x, tid = threadIdx.x;
    const float* row = w + (size_t)o * DIN;
    float v0 = row[tid];
    float mn = v0, mx = v0;
    for (int i = tid + 256; i < DIN; i += 256) {
        float v = row[i];
        mn = fminf(mn, v); mx = fmaxf(mx, v);
    }
    smin[tid] = mn; smax[tid] = mx;
    __syncthreads();
    for (int s = 128; s > 0; s >>= 1) {
        if (tid < s) {
            smin[tid] = fminf(smin[tid], smin[tid + s]);
            smax[tid] = fmaxf(smax[tid], smax[tid + s]);
        }
        __syncthreads();
    }
    if (tid == 0) {
        float wmin = smin[0], wmax = smax[0];
        float sw = (float)(NLEV - 1) / fmaxf(wmax - wmin, 1e-8f);
        float z  = sw * wmin;
        s_sw = sw; s_zp = z;
        zp[o]  = z;
        isw[o] = 1.0f / sw;
        float qb = rintf(bias[o] * sw * scal[0]);
        qbf[o] = fminf(fmaxf(qb, -(float)(NLEV + 1)), (float)NLEV);
    }
    __syncthreads();
    const float sw = s_sw, z = s_zp;
    unsigned char* qrow = qw + (size_t)o * DIN;
    for (int i = tid; i < DIN; i += 256)
        qrow[i] = (unsigned char)(int)rintf(sw * row[i] - z);
}

__global__ __launch_bounds__(256)
void k_xquant(const float* __restrict__ x, const float* __restrict__ scal,
              unsigned char* __restrict__ qx, int* __restrict__ rowsum) {
    __shared__ int ssum[256];
    const int t = blockIdx.x, tid = threadIdx.x;
    const float sx = scal[0];
    const float* row = x + (size_t)t * DIN;
    unsigned char* qrow = qx + (size_t)t * DIN;
    int s = 0;
    for (int i = tid; i < DIN; i += 256) {
        int q = (int)rintf(sx * row[i]);
        qrow[i] = (unsigned char)q;
        s += q;
    }
    ssum[tid] = s;
    __syncthreads();
    for (int st = 128; st > 0; st >>= 1) {
        if (tid < st) ssum[tid] += ssum[tid + st];
        __syncthreads();
    }
    if (tid == 0) rowsum[t] = ssum[0];
}

__global__ __launch_bounds__(256)
void k_gemm_iu8(const unsigned char* __restrict__ qx,
                const unsigned char* __restrict__ qw,
                const int*   __restrict__ rowsum,
                const float* __restrict__ zp,
                const float* __restrict__ isw,
                const float* __restrict__ qbf,
                const float* __restrict__ scal,
                float* __restrict__ out) {
    const int lane = threadIdx.x & 31;
    const int wave = threadIdx.x >> 5;
    const int lm   = lane & 15;
    const int hi   = lane >> 4;

    const int m0 = blockIdx.x * 128 + (wave >> 1) * 32;
    const int n0 = blockIdx.y * 128 + (wave & 1) * 64;

    v8i acc[2][4];
#pragma unroll
    for (int i = 0; i < 2; ++i)
#pragma unroll
        for (int j = 0; j < 4; ++j) acc[i][j] = (v8i)0;

    const unsigned char* aBase[2];
    aBase[0] = qx + (size_t)(m0 + lm) * DIN + hi * 16;
    aBase[1] = aBase[0] + (size_t)16 * DIN;
    const unsigned char* bBase[4];
#pragma unroll
    for (int j = 0; j < 4; ++j)
        bBase[j] = qw + (size_t)(n0 + j * 16 + lm) * DIN + hi * 16;

    for (int k0 = 0; k0 < DIN; k0 += 64) {
        v8i a[2], b[4];
#pragma unroll
        for (int i = 0; i < 2; ++i) {
            const unsigned char* p = aBase[i] + k0;
            uint64_t* f = reinterpret_cast<uint64_t*>(&a[i]);
            f[0] = *reinterpret_cast<const uint64_t*>(p);
            f[1] = *reinterpret_cast<const uint64_t*>(p + 8);
            f[2] = *reinterpret_cast<const uint64_t*>(p + 32);
            f[3] = *reinterpret_cast<const uint64_t*>(p + 40);
        }
#pragma unroll
        for (int j = 0; j < 4; ++j) {
            const unsigned char* p = bBase[j] + k0;
            v4i* f = reinterpret_cast<v4i*>(&b[j]);
            f[0] = *reinterpret_cast<const v4i*>(p);
            f[1] = *reinterpret_cast<const v4i*>(p + 32);
        }
        if (k0 + 64 < DIN) {
            __builtin_prefetch(aBase[0] + k0 + 64, 0, 1);
            __builtin_prefetch(bBase[0] + k0 + 64, 0, 1);
        }
#pragma unroll
        for (int i = 0; i < 2; ++i)
#pragma unroll
            for (int j = 0; j < 4; ++j)
                acc[i][j] = __builtin_amdgcn_wmma_i32_16x16x64_iu8(
                     false, a[i],  false, b[j],
                    acc[i][j],  false,  false);
    }

    const float inv_sx = scal[1];
    float zpc[4], swc[4], qbc[4];
#pragma unroll
    for (int j = 0; j < 4; ++j) {
        const int c = n0 + j * 16 + lm;
        zpc[j] = zp[c];
        swc[j] = isw[c] * inv_sx;
        qbc[j] = qbf[c];
    }
#pragma unroll
    for (int i = 0; i < 2; ++i) {
#pragma unroll
        for (int r = 0; r < 8; ++r) {
            const int row = m0 + i * 16 + hi * 8 + r;
            const float rs = (float)rowsum[row];
            float* orow = out + (size_t)row * DOUT;
#pragma unroll
            for (int j = 0; j < 4; ++j) {
                const int c = n0 + j * 16 + lm;
                float v = ((float)acc[i][j][r] + qbc[j]) + zpc[j] * rs;
                orow[c] = v * swc[j];
            }
        }
    }
}

extern "C" void kernel_launch(void* const* d_in, const int* in_sizes, int n_in,
                              void* d_out, int out_size, void* d_ws, size_t ws_size,
                              hipStream_t stream) {
    const float* x    = (const float*)d_in[0];
    const float* w    = (const float*)d_in[1];
    const float* bias = (const float*)d_in[2];
    (void)in_sizes; (void)n_in;

    char* ws = (char*)d_ws;
    float* scal    = (float*)(ws + 0);
    float* partial = (float*)(ws + 64);
    float* zp      = (float*)(ws + 4096);
    float* isw     = (float*)(ws + 4096 + 8192);
    float* qbf     = (float*)(ws + 4096 + 16384);
    int*   rowsum  = (int*)  (ws + 4096 + 24576);
    unsigned char* qw = (unsigned char*)(ws + 65536);
    unsigned char* qx = qw + (size_t)DOUT * DIN;
    float* out = (float*)d_out;
    (void)ws_size; (void)out_size;

    k_xmax_partial<<<256, 256, 0, stream>>>(x, partial, T_DIM * DIN);
    k_xscale      <<<1,   256, 0, stream>>>(partial, scal);
    k_wquant      <<<DOUT, 256, 0, stream>>>(w, bias, scal, zp, isw, qbf, qw);
    k_xquant      <<<T_DIM, 256, 0, stream>>>(x, scal, qx, rowsum);

    dim3 grid(T_DIM / 128, DOUT / 128);
    k_gemm_iu8<<<grid, 256, 0, stream>>>(qx, qw, rowsum, zp, isw, qbf, scal, out);
}
